// MySimpleRNN_90666759618801
// MI455X (gfx1250) — matrix operand order repaired and checked by reading, not compiled, not run on hardware
//
#include <hip/hip_runtime.h>
#include <math.h>

// ---------------------------------------------------------------------------
// MySimpleRNN on MI455X (gfx1250), wave32 + WMMA bf16.
//
// Phase 0: X -> bf16; W_xh -> WxT[u][d] bf16; W_hh -> WhT[n][k] bf16
//          (transposed so WMMA B-fragments are K-contiguous b128 loads).
// Phase 1: xw = X @ W_xh + b as a bf16 WMMA GEMM, f32 accum, stored in
//          WMMA C-fragment order [btile][t][ntile][lane][8] so the scan's
//          per-step seed is a single contiguous v8f load per lane.
// Phase 2: serial scan h_t = tanh(xw_t + h_{t-1} @ W_hh).
//          4 WGs (one 16-batch-row M-tile each), 16 waves x 2 N-tiles,
//          double-buffered bf16 h in LDS (one barrier/step), seed for t+1
//          prefetched during step t, WhT streamed from the 192MB L2.
// ---------------------------------------------------------------------------

typedef __attribute__((ext_vector_type(16))) __bf16 v16bf;
typedef __attribute__((ext_vector_type(8)))  __bf16 v8bf;
typedef __attribute__((ext_vector_type(8)))  float  v8f;

union ABFrag { v16bf v; v8bf h[2]; };

static constexpr int BB = 64;    // batch
static constexpr int TT = 1024;  // time steps
static constexpr int DD = 256;   // input dim
static constexpr int UU = 512;   // units

// xw fragment-layout index: btile in [0,4), t in [0,1024), ntile in [0,32),
// lane in [0,32), r in [0,8). Contiguous in r (32B per lane).
__device__ __forceinline__ size_t xw_idx(int bt, int t, int ntile, int lane) {
    return ((((size_t)bt * TT + t) * 32 + ntile) * 32 + lane) * 8;
}

// ---------------------------------------------------------------------------
// Phase 0 kernels: dtype conversion / transpose.
// ---------------------------------------------------------------------------
__global__ void k_cvt_bf16(const float* __restrict__ in, __bf16* __restrict__ out, int n) {
    int i = blockIdx.x * blockDim.x + threadIdx.x;
    if (i < n) out[i] = (__bf16)in[i];
}

// in: (R, C) row-major f32 ; out: (C, R) row-major bf16  (out[c][r] = in[r][c])
__global__ void k_transpose_cvt(const float* __restrict__ in, __bf16* __restrict__ out,
                                int R, int C) {
    int i = blockIdx.x * blockDim.x + threadIdx.x;
    if (i < R * C) {
        int r = i / C, c = i % C;
        out[(size_t)c * R + r] = (__bf16)in[i];
    }
}

// ---------------------------------------------------------------------------
// Phase 1: xw = X @ W_xh + b.  One wave = 16(batch) x 64(u) strip at one t.
// M-tile = 16 consecutive batch rows at a fixed t (matches the scan's tile),
// so accumulators are stored verbatim in fragment order.
// ---------------------------------------------------------------------------
__global__ void __launch_bounds__(256)
k_gemm_xw(const __bf16* __restrict__ Xb,   // (B, T, D) bf16
          const __bf16* __restrict__ WxT,  // (U, D)    bf16 = W_xh^T
          const float*  __restrict__ bias, // (U)
          float* __restrict__ xw_sw) {     // fragment layout, B*T*U floats
    const int lane  = threadIdx.x & 31;
    const int wave  = threadIdx.x >> 5;
    const int g     = blockIdx.x * 8 + wave;  // 32768 waves
    const int bt    = g >> 13;                // batch tile   [0,4)
    const int t     = (g >> 3) & (TT - 1);    // time step    [0,1024)
    const int ns    = g & 7;                  // 64-col strip [0,8)
    const int col0  = ns * 64;
    const int lhalf = lane >> 4;
    const int lm    = lane & 15;

    v8f acc[4] = {};

    for (int k = 0; k < DD; k += 32) {
        // A fragment (16x32 bf16): row = batch bt*16+lm at time t.
        ABFrag a;
        const __bf16* ap = Xb + ((size_t)(bt * 16 + lm) * TT + t) * DD + k + 8 * lhalf;
        a.h[0] = *(const v8bf*)(ap);
        a.h[1] = *(const v8bf*)(ap + 16);
#pragma unroll
        for (int j = 0; j < 4; ++j) {
            ABFrag b;
            const __bf16* bp = WxT + (size_t)(col0 + 16 * j + lm) * DD + k + 8 * lhalf;
            b.h[0] = *(const v8bf*)(bp);
            b.h[1] = *(const v8bf*)(bp + 16);
            acc[j] = __builtin_amdgcn_wmma_f32_16x16x32_bf16(
                false, a.v, false, b.v, (short)0, acc[j], false, false);
        }
    }

    // Add bias and store accumulators in fragment order (contiguous v8f).
#pragma unroll
    for (int j = 0; j < 4; ++j) {
        const float bv = bias[col0 + 16 * j + lm];
        v8f o;
#pragma unroll
        for (int r = 0; r < 8; ++r) o[r] = acc[j][r] + bv;
        *(v8f*)(xw_sw + xw_idx(bt, t, ns * 4 + j, lane)) = o;
    }
}

// ---------------------------------------------------------------------------
// Phase 2: serial scan. blockIdx.x = batch tile (16 rows). 16 waves, wave w
// owns columns [w*32, w*32+32). Double-buffered h in LDS; one barrier/step;
// next step's xw seed prefetched during the current step's K-loop.
// ---------------------------------------------------------------------------
__global__ void __launch_bounds__(512)
k_rnn_scan(const float* __restrict__ xw_sw, // fragment layout
           const __bf16* __restrict__ WhT,  // (U, U) bf16, WhT[n][k] = W_hh[k][n]
           float* __restrict__ out) {       // (B, T, U) f32
    __shared__ __bf16 hbuf[2][16 * UU];     // 2 x 16KB double buffer

    const int lane  = threadIdx.x & 31;
    const int wave  = threadIdx.x >> 5;     // 0..15
    const int lhalf = lane >> 4;
    const int lm    = lane & 15;
    const int bt    = blockIdx.x;           // batch tile
    const int row0  = bt * 16;
    const int col0  = wave * 32;

    // h_0 = 0 (only buffer 0 is read before being written)
    for (int i = threadIdx.x; i < 16 * UU; i += blockDim.x) hbuf[0][i] = (__bf16)0.0f;
    __syncthreads();

    // Preload seed for t = 0.
    v8f seed[2];
#pragma unroll
    for (int j = 0; j < 2; ++j)
        seed[j] = *(const v8f*)(xw_sw + xw_idx(bt, 0, wave * 2 + j, lane));

    for (int t = 0; t < TT; ++t) {
        const __bf16* hrd = hbuf[t & 1];
        __bf16*       hwr = hbuf[(t + 1) & 1];

        v8f acc[2] = { seed[0], seed[1] };

        // Prefetch next step's seed; latency hides under the K-loop below.
        if (t + 1 < TT) {
#pragma unroll
            for (int j = 0; j < 2; ++j)
                seed[j] = *(const v8f*)(xw_sw + xw_idx(bt, t + 1, wave * 2 + j, lane));
        }

        // acc += h_{t-1} @ W_hh  (K = 512 -> 16 WMMA K-steps)
        for (int k = 0; k < UU; k += 32) {
            ABFrag a;
            const __bf16* ap = hrd + lm * UU + k + 8 * lhalf;
            a.h[0] = *(const v8bf*)(ap);
            a.h[1] = *(const v8bf*)(ap + 16);
#pragma unroll
            for (int j = 0; j < 2; ++j) {
                ABFrag b;
                const __bf16* bp = WhT + (size_t)(col0 + 16 * j + lm) * UU + k + 8 * lhalf;
                b.h[0] = *(const v8bf*)(bp);
                b.h[1] = *(const v8bf*)(bp + 16);
                acc[j] = __builtin_amdgcn_wmma_f32_16x16x32_bf16(
                    false, a.v, false, b.v, (short)0, acc[j], false, false);
            }
        }

        // h_t = tanh(acc): f32 to out (64B row segments), bf16 to next buffer.
#pragma unroll
        for (int j = 0; j < 2; ++j) {
            const int n = col0 + 16 * j + lm;
#pragma unroll
            for (int r = 0; r < 8; ++r) {
                const int m  = r + 8 * lhalf;
                const float hv = tanhf(acc[j][r]);
                out[((size_t)(row0 + m) * TT + t) * UU + n] = hv;
                hwr[m * UU + n] = (__bf16)hv;
            }
        }
        __syncthreads();   // h_t published before step t+1 reads it
    }
}

// ---------------------------------------------------------------------------
// Host launcher. Workspace layout (bytes):
//   [0]          Xb    bf16  B*T*D = 33,554,432
//   [33,554,432] WxT   bf16  U*D   =    262,144
//   [33,816,576] WhT   bf16  U*U   =    524,288
//   [34,340,864] xw_sw f32   B*T*U = 134,217,728   (fragment layout)
// total ~168.6 MB
// ---------------------------------------------------------------------------
extern "C" void kernel_launch(void* const* d_in, const int* in_sizes, int n_in,
                              void* d_out, int out_size, void* d_ws, size_t ws_size,
                              hipStream_t stream) {
    const float* X    = (const float*)d_in[0];  // (B, T, D)
    const float* W_xh = (const float*)d_in[1];  // (D, U)
    const float* W_hh = (const float*)d_in[2];  // (U, U)
    const float* b_h  = (const float*)d_in[3];  // (U)
    float* out = (float*)d_out;                 // (B, T, U)

    char* ws = (char*)d_ws;
    __bf16* Xb    = (__bf16*)(ws);
    __bf16* WxT   = (__bf16*)(ws + 33554432);
    __bf16* WhT   = (__bf16*)(ws + 33816576);
    float*  xw_sw = (float*) (ws + 34340864);

    const int nX = BB * TT * DD;  // 16,777,216
    k_cvt_bf16<<<(nX + 255) / 256, 256, 0, stream>>>(X, Xb, nX);
    k_transpose_cvt<<<(DD * UU + 255) / 256, 256, 0, stream>>>(W_xh, WxT, DD, UU);
    k_transpose_cvt<<<(UU * UU + 255) / 256, 256, 0, stream>>>(W_hh, WhT, UU, UU);

    // 4 batch-tiles x 1024 t x 8 strips = 32768 waves = 4096 blocks x 8 waves.
    k_gemm_xw<<<4096, 256, 0, stream>>>(Xb, WxT, b_h, xw_sw);

    // One workgroup per 16 batch rows; serial T-loop inside.
    k_rnn_scan<<<BB / 16, 512, 0, stream>>>(xw_sw, WhT, out);
}
